// WT5GQA_70901320123189
// MI455X (gfx1250) — matrix operand order repaired and checked by reading, not compiled, not run on hardware
//
#include <hip/hip_runtime.h>
#include <math.h>

// ---------------------------------------------------------------- constants
constexpr int Bc  = 2;
constexpr int Sc  = 2048;
constexpr int Dc  = 1024;
constexpr int Hc  = 16;
constexpr int HDc = 64;
constexpr int Mtot = Bc * Sc;          // 4096 tokens
constexpr int BIASW = 2 * Sc - 1;      // 4095 relative offsets

typedef __attribute__((ext_vector_type(16))) __bf16 bf16x16;
typedef __attribute__((ext_vector_type(8)))  float  f32x8;

union Frag {
    bf16x16 v;
    uint4   q[2];
    unsigned short h[16];
};

__device__ __forceinline__ unsigned short f2bf(float f) {
    unsigned int u = __float_as_uint(f);
    unsigned int r = u + 0x7FFFu + ((u >> 16) & 1u);   // round-to-nearest-even
    return (unsigned short)(r >> 16);
}

__device__ __forceinline__ f32x8 wmma_bf16(const Frag& a, const Frag& b, f32x8 c) {
    return __builtin_amdgcn_wmma_f32_16x16x32_bf16(
        /*neg_a=*/false, a.v, /*neg_b=*/false, b.v,
        /*c_mod=*/(short)0, c, /*reuse_a=*/false, /*reuse_b=*/false);
}

// ---------------------------------------------------------------- prep kernels
__global__ void prep_x(const float* __restrict__ x, unsigned short* __restrict__ xb, int n) {
    int i = blockIdx.x * blockDim.x + threadIdx.x;
    if (i < n) xb[i] = f2bf(x[i]);
}

// GQA weight pooling (key_scale applied to K; V pooled RAW per reference bug),
// plus plain bf16 conversion of Wq / Wo.
__global__ void prep_w(const float* __restrict__ Wq, const float* __restrict__ Wk,
                       const float* __restrict__ Wv, const float* __restrict__ Wo,
                       const float* __restrict__ key_scale,
                       unsigned short* __restrict__ wq_b, unsigned short* __restrict__ wk_b,
                       unsigned short* __restrict__ wv_b, unsigned short* __restrict__ wo_b) {
    int i = blockIdx.x * blockDim.x + threadIdx.x;
    if (i >= Hc * HDc * Dc) return;
    int d  = i & (Dc - 1);
    int n  = i >> 10;            // output row 0..1023
    int hh = n >> 6;             // head
    int hd = n & 63;
    int kv = hh >> 2;            // group of 4 heads per kv head
    float sk = 0.f, sv = 0.f;
#pragma unroll
    for (int g = 0; g < 4; ++g) {
        int hs = kv * 4 + g;
        size_t off = ((size_t)(hs * HDc + hd)) * Dc + d;
        sk += Wk[off] * key_scale[hs];
        sv += Wv[off];
    }
    wk_b[i] = f2bf(sk);
    wv_b[i] = f2bf(sv);
    wq_b[i] = f2bf(Wq[i]);
    wo_b[i] = f2bf(Wo[i]);
}

// T5 bidirectional bucket -> bias LUT [H][4095], index = (kv - q) + 2047
__global__ void prep_bias(const float* __restrict__ bias_emb, float* __restrict__ biasTab) {
    int i = blockIdx.x * blockDim.x + threadIdx.x;
    if (i >= Hc * BIASW) return;
    int h   = i / BIASW;
    int rel = (i % BIASW) - (Sc - 1);
    int bucket = (rel > 0) ? 16 : 0;
    int rela   = rel < 0 ? -rel : rel;
    if (rela < 8) {
        bucket += rela;
    } else {
        float rf = (float)rela;
        int large = 8 + (int)(logf(rf * 0.125f) * (8.0f / logf(16.0f)));
        if (large > 15) large = 15;
        bucket += large;
    }
    biasTab[i] = bias_emb[bucket * Hc + h];
}

// ---------------------------------------------------------------- bf16 GEMM
// C[m,n] = sum_k A[m,k] * W[n,k];   A:[4096,1024] bf16, W:[1024,1024] bf16
// mode 0: store bf16 to [B,H,S,HD]   (Q / K)
// mode 2: store bf16 to [B,H,HD,S]   (V transposed)
// mode 3: store fp32 to [M,1024]     (final output)
__global__ __launch_bounds__(256) void gemm_bf16(const unsigned short* __restrict__ A,
                                                 const unsigned short* __restrict__ W,
                                                 void* __restrict__ Out, int mode) {
    constexpr int K = 1024, N = 1024;
    const int lane  = threadIdx.x & 31;
    const int wave  = threadIdx.x >> 5;                 // 0..7
    const int mbase = blockIdx.x * 128 + (wave >> 1) * 32;
    const int nbase = blockIdx.y * 64  + (wave & 1) * 32;
    const int lm = lane & 15;
    const int kh = lane >> 4;

    f32x8 acc[2][2] = {};
    for (int k0 = 0; k0 < K; k0 += 32) {
        Frag a[2], b[2];
#pragma unroll
        for (int t = 0; t < 2; ++t) {
            // A fragment: lane holds row (mbase+t*16+lm); 16 contiguous K per half-wave
            const unsigned short* pa = A + (size_t)(mbase + t * 16 + lm) * K + k0 + kh * 16;
            a[t].q[0] = *(const uint4*)pa;
            a[t].q[1] = *(const uint4*)(pa + 8);
            // B fragment: lane holds column (nbase+t*16+lm); 16 contiguous K per half-wave
            const unsigned short* pb = W + (size_t)(nbase + t * 16 + lm) * K + k0 + kh * 16;
            b[t].q[0] = *(const uint4*)pb;
            b[t].q[1] = *(const uint4*)(pb + 8);
        }
#pragma unroll
        for (int mt = 0; mt < 2; ++mt)
#pragma unroll
            for (int nt = 0; nt < 2; ++nt)
                acc[mt][nt] = wmma_bf16(a[mt], b[nt], acc[mt][nt]);
    }

#pragma unroll
    for (int mt = 0; mt < 2; ++mt)
#pragma unroll
        for (int nt = 0; nt < 2; ++nt)
#pragma unroll
            for (int r = 0; r < 8; ++r) {
                int m = mbase + mt * 16 + kh * 8 + r;   // C layout: row = halfwave*8 + vgpr
                int n = nbase + nt * 16 + lm;           //           col = lane&15
                float val = acc[mt][nt][r];
                if (mode == 3) {
                    ((float*)Out)[(size_t)m * N + n] = val;
                } else {
                    int bb = m >> 11, s = m & (Sc - 1);
                    int hh = n >> 6,  hd = n & 63;
                    size_t off = (mode == 2)
                        ? (((size_t)(bb * Hc + hh)) * HDc + hd) * Sc + s
                        : (((size_t)(bb * Hc + hh)) * Sc + s) * HDc + hd;
                    ((unsigned short*)Out)[off] = f2bf(val);
                }
            }
}

// ---------------------------------------------------------------- flash attention
// One wave per (b, h, 16-row q tile). kv tiles of 32. Online softmax in fp32.
__global__ __launch_bounds__(256) void attn(const unsigned short* __restrict__ Qb,
                                            const unsigned short* __restrict__ Kb,
                                            const unsigned short* __restrict__ Vt,
                                            const float* __restrict__ biasTab,
                                            unsigned short* __restrict__ AOb) {
    __shared__ __align__(16) unsigned short ldsP[8][16 * 32];  // per-wave P staging
    const int lane = threadIdx.x & 31;
    const int wave = threadIdx.x >> 5;
    const int wid  = blockIdx.x * 8 + wave;      // 0..4095
    const int qt   = wid & 127;
    const int h    = (wid >> 7) & (Hc - 1);
    const int b    = wid >> 11;
    const int qbase = qt * 16;
    const int lm = lane & 15;
    const int kh = lane >> 4;

    // persistent Q A-fragments over HD=64 (two 16x32 slabs)
    Frag aq[2];
    {
        const unsigned short* qrow = Qb + (((size_t)(b * Hc + h)) * Sc + qbase + lm) * HDc;
#pragma unroll
        for (int t = 0; t < 2; ++t) {
            aq[t].q[0] = *(const uint4*)(qrow + t * 32 + kh * 16);
            aq[t].q[1] = *(const uint4*)(qrow + t * 32 + kh * 16 + 8);
        }
    }

    f32x8 o[4] = {};
    float mrow[8], lrow[8];
#pragma unroll
    for (int r = 0; r < 8; ++r) { mrow[r] = -INFINITY; lrow[r] = 0.f; }

    const unsigned short* Kbase = Kb + ((size_t)(b * Hc + h)) * Sc * HDc;
    const unsigned short* Vbase = Vt + ((size_t)(b * Hc + h)) * HDc * Sc;
    const float* btab = biasTab + h * BIASW;

    for (int j = 0; j < Sc / 32; ++j) {
        const int kvbase = j * 32;

        // scores: S[16 x 32] as two 16x16 C tiles
        f32x8 cs[2];
#pragma unroll
        for (int c = 0; c < 2; ++c) {
            const unsigned short* krow = Kbase + (size_t)(kvbase + c * 16 + lm) * HDc;
            f32x8 z = {};
            Frag kb;
            kb.q[0] = *(const uint4*)(krow + kh * 16);
            kb.q[1] = *(const uint4*)(krow + kh * 16 + 8);
            z = wmma_bf16(aq[0], kb, z);
            kb.q[0] = *(const uint4*)(krow + 32 + kh * 16);
            kb.q[1] = *(const uint4*)(krow + 32 + kh * 16 + 8);
            z = wmma_bf16(aq[1], kb, z);
            cs[c] = z;
        }

        // bias + online softmax (row r lives in one 16-lane half-wave)
#pragma unroll
        for (int r = 0; r < 8; ++r) {
            int qrow_i = qbase + kh * 8 + r;
            int col0   = kvbase + lm;
            float s0 = cs[0][r] + btab[col0 - qrow_i + (Sc - 1)];
            float s1 = cs[1][r] + btab[col0 + 16 - qrow_i + (Sc - 1)];
            float mx = fmaxf(s0, s1);
#pragma unroll
            for (int msk = 1; msk < 16; msk <<= 1)
                mx = fmaxf(mx, __shfl_xor(mx, msk, 32));
            float mnew = fmaxf(mrow[r], mx);
            float p0 = __expf(s0 - mnew);
            float p1 = __expf(s1 - mnew);
            float rs = p0 + p1;
#pragma unroll
            for (int msk = 1; msk < 16; msk <<= 1)
                rs += __shfl_xor(rs, msk, 32);
            float scale = __expf(mrow[r] - mnew);
            lrow[r] = lrow[r] * scale + rs;
            mrow[r] = mnew;
#pragma unroll
            for (int t = 0; t < 4; ++t) o[t][r] *= scale;
            // C-layout -> LDS row-major [16][32] for A-fragment reload
            int row = kh * 8 + r;
            ldsP[wave][row * 32 + lm]      = f2bf(p0);
            ldsP[wave][row * 32 + 16 + lm] = f2bf(p1);
        }
        asm volatile("s_wait_dscnt 0" ::: "memory");

        Frag ap;   // P as 16x32 A fragment
        ap.q[0] = *(const uint4*)(&ldsP[wave][lm * 32 + kh * 16]);
        ap.q[1] = *(const uint4*)(&ldsP[wave][lm * 32 + kh * 16 + 8]);

        // O[16x64] += P[16x32] @ V[32x64]  (V transposed: rows contiguous in kv)
#pragma unroll
        for (int t = 0; t < 4; ++t) {
            const unsigned short* vrow = Vbase + (size_t)(t * 16 + lm) * Sc + kvbase + kh * 16;
            Frag vb;
            vb.q[0] = *(const uint4*)vrow;
            vb.q[1] = *(const uint4*)(vrow + 8);
            o[t] = wmma_bf16(ap, vb, o[t]);
        }
    }

    // normalize + store bf16 attn output [B,S,H*HD]
#pragma unroll
    for (int r = 0; r < 8; ++r) {
        float inv = 1.0f / lrow[r];
        int s = qbase + kh * 8 + r;
        size_t base = ((size_t)b * Sc + s) * (Hc * HDc) + h * HDc;
#pragma unroll
        for (int t = 0; t < 4; ++t)
            AOb[base + t * 16 + lm] = f2bf(o[t][r] * inv);
    }
}

// ---------------------------------------------------------------- launcher
extern "C" void kernel_launch(void* const* d_in, const int* in_sizes, int n_in,
                              void* d_out, int out_size, void* d_ws, size_t ws_size,
                              hipStream_t stream) {
    const float* x    = (const float*)d_in[0];
    const float* Wq   = (const float*)d_in[1];
    const float* Wk   = (const float*)d_in[2];
    const float* Wv   = (const float*)d_in[3];
    const float* Wo   = (const float*)d_in[4];
    const float* ksc  = (const float*)d_in[5];
    // d_in[6] = value_scale: unused (reference pools raw V weight)
    const float* bemb = (const float*)d_in[7];

    size_t off = 0;
    auto carve = [&](size_t bytes) {
        void* p = (char*)d_ws + off;
        off += (bytes + 255) & ~(size_t)255;
        return p;
    };
    const size_t nX = (size_t)Mtot * Dc;          // 4M
    const size_t nW = (size_t)Dc * Dc;            // 1M
    unsigned short* xb   = (unsigned short*)carve(nX * 2);
    unsigned short* wq_b = (unsigned short*)carve(nW * 2);
    unsigned short* wk_b = (unsigned short*)carve(nW * 2);
    unsigned short* wv_b = (unsigned short*)carve(nW * 2);
    unsigned short* wo_b = (unsigned short*)carve(nW * 2);
    unsigned short* Qb   = (unsigned short*)carve(nX * 2);
    unsigned short* Kb   = (unsigned short*)carve(nX * 2);
    unsigned short* Vt   = (unsigned short*)carve(nX * 2);
    unsigned short* AOb  = (unsigned short*)carve(nX * 2);
    float*          bTab = (float*)carve((size_t)Hc * BIASW * 4);

    prep_x   <<<(int)((nX + 255) / 256), 256, 0, stream>>>(x, xb, (int)nX);
    prep_w   <<<(int)((nW + 255) / 256), 256, 0, stream>>>(Wq, Wk, Wv, Wo, ksc,
                                                           wq_b, wk_b, wv_b, wo_b);
    prep_bias<<<(Hc * BIASW + 255) / 256, 256, 0, stream>>>(bemb, bTab);

    dim3 gGemm(Mtot / 128, Dc / 64);   // 32 x 16
    gemm_bf16<<<gGemm, 256, 0, stream>>>(xb, wq_b, Qb, 0);
    gemm_bf16<<<gGemm, 256, 0, stream>>>(xb, wk_b, Kb, 0);
    gemm_bf16<<<gGemm, 256, 0, stream>>>(xb, wv_b, Vt, 2);

    attn<<<(Bc * Hc * (Sc / 16)) / 8, 256, 0, stream>>>(Qb, Kb, Vt, bTab, AOb);

    gemm_bf16<<<gGemm, 256, 0, stream>>>(AOb, wo_b, d_out, 3);
}
